// NystromAttention_3032246911013
// MI455X (gfx1250) — matrix operand order repaired and checked by reading, not compiled, not run on hardware
//
#include <hip/hip_runtime.h>
#include <hip/hip_bf16.h>

typedef __bf16 bf16_t;
typedef __attribute__((ext_vector_type(8)))  __bf16 v8bf;
typedef __attribute__((ext_vector_type(16))) __bf16 v16bf;
typedef __attribute__((ext_vector_type(8)))  float  v8f;

// ---------------------------------------------------------------- WMMA GEMM
__device__ inline v16bf cat8(v8bf lo, v8bf hi) {
  v16bf r;
#pragma unroll
  for (int i = 0; i < 8; ++i) { r[i] = lo[i]; r[8 + i] = hi[i]; }
  return r;
}

__device__ inline v8f wmma_bf16(v16bf a, v16bf b, v8f c) {
  return __builtin_amdgcn_wmma_f32_16x16x32_bf16(false, a, false, b, (short)0, c,
                                                 false, false);
}

// C[M,N] = scale * A[M,K] * B[N,K]^T   (NT gemm, bf16 in, f32 accumulate)
// Epilogue (all optional):
//   outF [m,n] = val + bias[n]                     (f32)
//   outRM[m,n] = bf16(val)                         (row-major bf16)
//   outTR[n,m] = bf16(trsign*val + (m==n)?trdiag)  (transposed bf16, fused aI-X)
// Block = (32,4): 4 waves, each wave a 32x32 tile, block tile 64x64.
__global__ __launch_bounds__(128) void gemm_nt(
    const bf16_t* __restrict__ A, const bf16_t* __restrict__ B,
    int K, int lda, int ldb, long long sA, long long sB,
    float* __restrict__ outF, long long sF, int ldf, const float* __restrict__ bias,
    bf16_t* __restrict__ outRM, long long sRM, int ldrm,
    bf16_t* __restrict__ outTR, long long sTR, int ldtr,
    float trdiag, float trsign, float scale) {
  const int lane = threadIdx.x;
  const int wv   = threadIdx.y;
  const int half = lane >> 4;
  const int l16  = lane & 15;
  const int z    = blockIdx.z;
  A += (long long)z * sA;
  B += (long long)z * sB;
  const int tm = blockIdx.x * 64 + (wv & 1) * 32;
  const int tn = blockIdx.y * 64 + (wv >> 1) * 32;

  v8f c00 = {}, c01 = {}, c10 = {}, c11 = {};

  // CDNA5 16-bit A 16x32 layout: lanes 0-15 hold K[0..7],[16..23]; lanes 16-31
  // hold K[8..15],[24..31].  B 32x16: same K order as A.
  const bf16_t* a0p = A + (long long)(tm + l16) * lda + half * 8;
  const bf16_t* a1p = a0p + 16LL * lda;
  const bf16_t* b0p = B + (long long)(tn + l16) * ldb + half * 8;
  const bf16_t* b1p = b0p + 16LL * ldb;

  for (int k0 = 0; k0 < K; k0 += 32) {
    v16bf a0 = cat8(*(const v8bf*)(a0p + k0), *(const v8bf*)(a0p + k0 + 16));
    v16bf a1 = cat8(*(const v8bf*)(a1p + k0), *(const v8bf*)(a1p + k0 + 16));
    v16bf b0 = cat8(*(const v8bf*)(b0p + k0), *(const v8bf*)(b0p + k0 + 16));
    v16bf b1 = cat8(*(const v8bf*)(b1p + k0), *(const v8bf*)(b1p + k0 + 16));
    c00 = wmma_bf16(a0, b0, c00);
    c01 = wmma_bf16(a0, b1, c01);
    c10 = wmma_bf16(a1, b0, c10);
    c11 = wmma_bf16(a1, b1, c11);
  }

  float*  oF  = outF  ? outF  + (long long)z * sF  : nullptr;
  bf16_t* oRM = outRM ? outRM + (long long)z * sRM : nullptr;
  bf16_t* oTR = outTR ? outTR + (long long)z * sTR : nullptr;

  auto emit = [&](v8f c, int sm, int sn) {
#pragma unroll
    for (int r = 0; r < 8; ++r) {
      // C layout: lane<16 -> M=r, lane>=16 -> M=8+r ; N = lane&15
      int   m = tm + sm + half * 8 + r;
      int   n = tn + sn + l16;
      float v = scale * c[r];
      if (oF)  oF[(long long)m * ldf + n] = v + (bias ? bias[n] : 0.f);
      if (oRM) oRM[(long long)m * ldrm + n] = (bf16_t)v;
      if (oTR) {
        float tv = trsign * v + ((m == n) ? trdiag : 0.f);
        oTR[(long long)n * ldtr + m] = (bf16_t)tv;
      }
    }
  };
  emit(c00, 0, 0); emit(c01, 0, 16); emit(c10, 16, 0); emit(c11, 16, 16);
}

// ---------------------------------------------------------------- softmax
__global__ __launch_bounds__(256) void softmax_rows(
    const float* __restrict__ in, float* __restrict__ outF,
    bf16_t* __restrict__ outB, int C) {
  const int row = blockIdx.x;
  const int t   = threadIdx.x;
  const float* ir = in + (long long)row * C;
  __shared__ float red[256];
  float vals[16];
  const int cnt = C >> 8;  // C is a multiple of 256, C<=4096
  float mx = -3.4e38f;
  for (int i = 0; i < cnt; ++i) {
    float x = ir[t + (i << 8)];
    vals[i] = x;
    mx = fmaxf(mx, x);
  }
  red[t] = mx; __syncthreads();
  for (int s = 128; s > 0; s >>= 1) { if (t < s) red[t] = fmaxf(red[t], red[t + s]); __syncthreads(); }
  float rmax = red[0]; __syncthreads();
  float sum = 0.f;
  for (int i = 0; i < cnt; ++i) { float e = expf(vals[i] - rmax); vals[i] = e; sum += e; }
  red[t] = sum; __syncthreads();
  for (int s = 128; s > 0; s >>= 1) { if (t < s) red[t] += red[t + s]; __syncthreads(); }
  float inv = 1.f / red[0];
  for (int i = 0; i < cnt; ++i) {
    int   j = t + (i << 8);
    float v = vals[i] * inv;
    if (outF) outF[(long long)row * C + j] = v;
    if (outB) outB[(long long)row * C + j] = (bf16_t)v;
  }
}

// -------------------------------------------------- elementwise / transposes
__global__ void f32_to_bf16_k(const float* __restrict__ in, bf16_t* __restrict__ o,
                              long long n) {
  long long i = (long long)blockIdx.x * 256 + threadIdx.x;
  if (i < n) o[i] = (bf16_t)in[i];
}

__global__ void transpose_bf16_k(const float* __restrict__ in, bf16_t* __restrict__ o,
                                 int R, int C) {
  long long i = (long long)blockIdx.x * 256 + threadIdx.x;
  if (i >= (long long)R * C) return;
  int c = (int)(i % C), r = (int)(i / C);
  o[(long long)c * R + r] = (bf16_t)in[i];
}

// qkv_f32 [b,n,3072] -> q_bf/k_bf [bh,n,64] (q scaled), v_t [bh,64,n]
__global__ void split_heads_k(const float* __restrict__ qkv, bf16_t* __restrict__ q,
                              bf16_t* __restrict__ k, bf16_t* __restrict__ vt) {
  long long idx = (long long)blockIdx.x * 256 + threadIdx.x;
  if (idx >= 3LL * 4 * 16 * 4096 * 64) return;
  int dd = (int)(idx & 63); long long r = idx >> 6;
  int nn = (int)(r & 4095); r >>= 12;
  int h  = (int)(r & 15);   r >>= 4;
  int bb = (int)(r & 3);    r >>= 2;
  int which = (int)r;
  float v = qkv[((long long)(bb * 4096 + nn)) * 3072 + which * 1024 + h * 64 + dd];
  long long bh = bb * 16 + h;
  if (which == 0)      q[(bh * 4096 + nn) * 64 + dd] = (bf16_t)(v * 0.125f);
  else if (which == 1) k[(bh * 4096 + nn) * 64 + dd] = (bf16_t)v;
  else                 vt[(bh * 64 + dd) * 4096LL + nn] = (bf16_t)v;
}

// mean over 16 consecutive tokens -> landmarks [bh,256,64]
__global__ void landmarks_k(const bf16_t* __restrict__ q, const bf16_t* __restrict__ k,
                            bf16_t* __restrict__ ql, bf16_t* __restrict__ kl) {
  long long idx = (long long)blockIdx.x * 256 + threadIdx.x;
  if (idx >= 2LL * 64 * 256 * 64) return;
  int dd = (int)(idx & 63); long long r = idx >> 6;
  int m  = (int)(r & 255);  r >>= 8;
  int bh = (int)(r & 63);   r >>= 6;
  int which = (int)r;
  const bf16_t* src = which ? k : q;
  long long base = ((long long)bh * 4096 + m * 16) * 64 + dd;
  float s = 0.f;
#pragma unroll
  for (int t = 0; t < 16; ++t) s += (float)src[base + t * 64];
  (which ? kl : ql)[((long long)bh * 256 + m) * 64 + dd] = (bf16_t)(s * 0.0625f);
}

__global__ void init_scal_k(float* s) { s[0] = 0.f; s[1] = 0.f; }

// global max of row/col abs-sums of softmaxed attn2
__global__ __launch_bounds__(256) void attn2_scale_k(const float* __restrict__ a,
                                                     float* __restrict__ scal) {
  int bh = blockIdx.x, t = threadIdx.x;
  const float* ab = a + (long long)bh * 65536;
  float cs = 0.f, rs = 0.f;
  for (int j = 0; j < 256; ++j) {
    cs += fabsf(ab[t * 256 + j]);   // col = sum over last axis
    rs += fabsf(ab[j * 256 + t]);   // row = sum over axis -2
  }
  __shared__ float sm[256];
  sm[t] = cs; __syncthreads();
  for (int s = 128; s > 0; s >>= 1) { if (t < s) sm[t] = fmaxf(sm[t], sm[t + s]); __syncthreads(); }
  if (!t) atomicMax((unsigned int*)&scal[0], __float_as_uint(sm[0]));
  __syncthreads();
  sm[t] = rs; __syncthreads();
  for (int s = 128; s > 0; s >>= 1) { if (t < s) sm[t] = fmaxf(sm[t], sm[t + s]); __syncthreads(); }
  if (!t) atomicMax((unsigned int*)&scal[1], __float_as_uint(sm[0]));
}

// z0 = a^T / (max_col * max_row)  ->  z_rm and z_t (bf16)
__global__ void z0_init_k(const float* __restrict__ a, const float* __restrict__ scal,
                          bf16_t* __restrict__ zrm, bf16_t* __restrict__ zt) {
  long long idx = (long long)blockIdx.x * 256 + threadIdx.x;
  if (idx >= 64LL * 256 * 256) return;
  int j = (int)(idx & 255); long long r = idx >> 8;
  int i = (int)(r & 255);
  int bh = (int)(r >> 8);
  float s = 1.f / (scal[0] * scal[1]);
  const float* ab = a + (long long)bh * 65536;
  zrm[idx] = (bf16_t)(ab[j * 256 + i] * s);  // z[i][j]   = a[j][i]/s
  zt[idx]  = (bf16_t)(ab[i * 256 + j] * s);  // z^T[i][j] = a[i][j]/s
}

// depthwise conv (K=13, corr, zero pad) + residual + merge heads -> y_bf [b*n,1024]
__global__ void conv_residual_k(const float* __restrict__ o, const float* __restrict__ wc,
                                bf16_t* __restrict__ y) {
  long long idx = (long long)blockIdx.x * 256 + threadIdx.x;
  if (idx >= 64LL * 4096 * 64) return;
  int dd = (int)(idx & 63); long long r = idx >> 6;
  int i  = (int)(r & 4095);
  int bh = (int)(r >> 12);
  int h = bh & 15, bb = bh >> 4;
  const float* base = o + ((long long)bh * 4096) * 64 + dd;
  float v0 = base[(long long)i * 64];
  float acc = 0.f;
#pragma unroll
  for (int t = 0; t < 13; ++t) {
    int j = i + t - 6;
    if (j >= 0 && j < 4096) acc += wc[h * 13 + t] * base[(long long)j * 64];
  }
  y[((long long)(bb * 4096 + i)) * 1024 + h * 64 + dd] = (bf16_t)(v0 + acc);
}

// ---------------------------------------------------------------- host side
static inline void launch_gemm(hipStream_t s, dim3 grid,
    const bf16_t* A, const bf16_t* B, int K, int lda, int ldb,
    long long sA, long long sB,
    float* outF, long long sF, int ldf, const float* bias,
    bf16_t* outRM, long long sRM, int ldrm,
    bf16_t* outTR, long long sTR, int ldtr,
    float trdiag, float trsign, float scale) {
  gemm_nt<<<grid, dim3(32, 4), 0, s>>>(A, B, K, lda, ldb, sA, sB, outF, sF, ldf,
                                       bias, outRM, sRM, ldrm, outTR, sTR, ldtr,
                                       trdiag, trsign, scale);
}

extern "C" void kernel_launch(void* const* d_in, const int* in_sizes, int n_in,
                              void* d_out, int out_size, void* d_ws, size_t ws_size,
                              hipStream_t stream) {
  (void)in_sizes; (void)n_in; (void)out_size; (void)ws_size;
  const float* x      = (const float*)d_in[0];
  const float* w_qkv  = (const float*)d_in[1];
  const float* w_conv = (const float*)d_in[2];
  const float* w_fc   = (const float*)d_in[3];
  const float* b_fc   = (const float*)d_in[4];
  float* out = (float*)d_out;

  char* w = (char*)d_ws;
  size_t off = 0;
  auto alloc = [&](size_t bytes) -> char* {
    char* p = w + off;
    off += (bytes + 255) & ~(size_t)255;
    return p;
  };

  // big scratch, reused: qkv_f32 (201MB) -> sim1/sim3 f32 (268MB) -> out_f32 + y_bf
  float*  big    = (float*) alloc(268435456ull);
  bf16_t* x_bf   = (bf16_t*)alloc(33554432ull);
  bf16_t* wqkv_t = (bf16_t*)alloc(6291456ull);
  bf16_t* wfc_t  = (bf16_t*)alloc(2097152ull);
  bf16_t* q_bf   = (bf16_t*)alloc(33554432ull);
  bf16_t* k_bf   = (bf16_t*)alloc(33554432ull);
  bf16_t* v_t    = (bf16_t*)alloc(33554432ull);
  bf16_t* ql     = (bf16_t*)alloc(2097152ull);
  bf16_t* kl     = (bf16_t*)alloc(2097152ull);
  bf16_t* attn1  = (bf16_t*)alloc(134217728ull);
  bf16_t* attn3  = (bf16_t*)alloc(134217728ull);
  float*  attn2f = (float*) alloc(16777216ull);
  bf16_t* a_rm   = (bf16_t*)alloc(8388608ull);
  bf16_t* az     = (bf16_t*)alloc(8388608ull);
  bf16_t* e7     = (bf16_t*)alloc(8388608ull);
  bf16_t* e15    = (bf16_t*)alloc(8388608ull);
  bf16_t* e13    = (bf16_t*)alloc(8388608ull);
  bf16_t* z_rm[2] = {(bf16_t*)alloc(8388608ull), (bf16_t*)alloc(8388608ull)};
  bf16_t* z_t[2]  = {(bf16_t*)alloc(8388608ull), (bf16_t*)alloc(8388608ull)};
  bf16_t* kv_t   = (bf16_t*)alloc(2097152ull);
  bf16_t* zkv_t  = (bf16_t*)alloc(2097152ull);
  float*  scal   = (float*) alloc(256ull);

  float*  qkv_f32 = big;                                 // [16384,3072]
  float*  sim_f32 = big;                                 // [*,<=268MB]
  float*  out_f32 = big;                                 // [64,4096,64]
  bf16_t* y_bf    = (bf16_t*)((char*)big + 134217728ull);// [16384,1024]

  // 1) precision conversion / weight transposes
  f32_to_bf16_k<<<65536, 256, 0, stream>>>(x, x_bf, 16777216LL);
  transpose_bf16_k<<<12288, 256, 0, stream>>>(w_qkv, wqkv_t, 1024, 3072);
  transpose_bf16_k<<<4096, 256, 0, stream>>>(w_fc, wfc_t, 1024, 1024);
  init_scal_k<<<1, 1, 0, stream>>>(scal);

  // 2) qkv = x @ w_qkv     [16384,3072]
  launch_gemm(stream, dim3(256, 48, 1), x_bf, wqkv_t, 1024, 1024, 1024, 0, 0,
              qkv_f32, 0, 3072, nullptr, nullptr, 0, 0, nullptr, 0, 0, 0.f, 1.f, 1.f);

  // 3) split heads (+scale q, v stored transposed)
  split_heads_k<<<196608, 256, 0, stream>>>(qkv_f32, q_bf, k_bf, v_t);

  // 4) landmarks (means of 16 tokens)
  landmarks_k<<<8192, 256, 0, stream>>>(q_bf, k_bf, ql, kl);

  // 5) sim1 = q @ k_l^T  [bh,4096,256], softmax -> attn1 (bf16)
  launch_gemm(stream, dim3(64, 4, 64), q_bf, kl, 64, 64, 64, 262144, 16384,
              sim_f32, 1048576, 256, nullptr, nullptr, 0, 0, nullptr, 0, 0, 0.f, 1.f, 1.f);
  softmax_rows<<<262144, 256, 0, stream>>>(sim_f32, nullptr, attn1, 256);

  // 6) sim2 = q_l @ k_l^T [bh,256,256], softmax -> attn2 (f32 in-place + bf16)
  launch_gemm(stream, dim3(4, 4, 64), ql, kl, 64, 64, 64, 16384, 16384,
              attn2f, 65536, 256, nullptr, nullptr, 0, 0, nullptr, 0, 0, 0.f, 1.f, 1.f);
  softmax_rows<<<16384, 256, 0, stream>>>(attn2f, attn2f, a_rm, 256);

  // 7) z0 = attn2^T / (max(colsum)*max(rowsum))
  attn2_scale_k<<<64, 256, 0, stream>>>(attn2f, scal);
  z0_init_k<<<16384, 256, 0, stream>>>(attn2f, scal, z_rm[0], z_t[0]);

  // 8) sim3 = q_l @ k^T  [bh,256,4096], softmax -> attn3 (bf16)
  launch_gemm(stream, dim3(4, 64, 64), ql, k_bf, 64, 64, 64, 16384, 262144,
              sim_f32, 1048576, 4096, nullptr, nullptr, 0, 0, nullptr, 0, 0, 0.f, 1.f, 1.f);
  softmax_rows<<<16384, 256, 0, stream>>>(sim_f32, nullptr, attn3, 4096);

  // 9) Moore-Penrose: z = 0.25 z (13I - az(15I - az(7I - az)))
  dim3 gmp(4, 4, 64);
  for (int it = 0; it < 6; ++it) {
    int p = it & 1;
    // az = a@z ; also e7 = (7I - az)^T
    launch_gemm(stream, gmp, a_rm, z_t[p], 256, 256, 256, 65536, 65536,
                nullptr, 0, 1, nullptr, az, 65536, 256, e7, 65536, 256, 7.f, -1.f, 1.f);
    // e15 = (15I - az@e7^T... ) stored transposed: t3 = az@(7I-az); e15 = (15I-t3)^T
    launch_gemm(stream, gmp, az, e7, 256, 256, 256, 65536, 65536,
                nullptr, 0, 1, nullptr, nullptr, 0, 0, e15, 65536, 256, 15.f, -1.f, 1.f);
    // e13 = (13I - az@(15I-t3))^T
    launch_gemm(stream, gmp, az, e15, 256, 256, 256, 65536, 65536,
                nullptr, 0, 1, nullptr, nullptr, 0, 0, e13, 65536, 256, 13.f, -1.f, 1.f);
    // z_new = 0.25 * z @ (13I - ...)  (write both row-major and transposed)
    launch_gemm(stream, gmp, z_rm[p], e13, 256, 256, 256, 65536, 65536,
                nullptr, 0, 1, nullptr, z_rm[1 - p], 65536, 256,
                z_t[1 - p], 65536, 256, 0.f, 1.f, 0.25f);
  }
  // after 6 iterations final z lives in buffer 0

  // 10) kv = attn3 @ v  [bh,256,64]  -> stored transposed kv_t [64,256]
  launch_gemm(stream, dim3(4, 1, 64), attn3, v_t, 4096, 4096, 4096, 1048576, 262144,
              nullptr, 0, 1, nullptr, nullptr, 0, 0, kv_t, 16384, 256, 0.f, 1.f, 1.f);

  // 11) zkv = z @ kv  [bh,256,64] -> stored transposed zkv_t [64,256]
  launch_gemm(stream, dim3(4, 1, 64), z_rm[0], kv_t, 256, 256, 256, 65536, 16384,
              nullptr, 0, 1, nullptr, nullptr, 0, 0, zkv_t, 16384, 256, 0.f, 1.f, 1.f);

  // 12) out = attn1 @ zkv  [bh,4096,64] (f32)
  launch_gemm(stream, dim3(64, 1, 64), attn1, zkv_t, 256, 256, 256, 1048576, 16384,
              out_f32, 262144, 64, nullptr, nullptr, 0, 0, nullptr, 0, 0, 0.f, 1.f, 1.f);

  // 13) depthwise conv residual + merge heads -> y_bf [16384,1024]
  conv_residual_k<<<65536, 256, 0, stream>>>(out_f32, w_conv, y_bf);

  // 14) final fc: out = y @ w_fc + b_fc  [16384,1024] f32 -> d_out
  launch_gemm(stream, dim3(256, 16, 1), y_bf, wfc_t, 1024, 1024, 1024, 0, 0,
              out, 0, 1024, b_fc, nullptr, 0, 0, nullptr, 0, 0, 0.f, 1.f, 1.f);
}
